// RobustGeoSNN_65420941853065
// MI455X (gfx1250) — matrix operand order repaired and checked by reading, not compiled, not run on hardware
//
#include <hip/hip_runtime.h>
#include <hip/hip_bf16.h>

// ---------------------------------------------------------------------------
// RobustGeoSNN — CDNA5 (gfx1250) implementation
//   conv1d(+BN+ReLU) -> LIF SNN scan (T=1000) -> TDA MLP + classifier
// All big GEMMs via v_wmma_f32_16x16x32_f16 (f16 in, f32 accumulate).
// ---------------------------------------------------------------------------

typedef __attribute__((ext_vector_type(16))) _Float16 v16h;
typedef __attribute__((ext_vector_type(8)))  _Float16 v8h;
typedef __attribute__((ext_vector_type(8)))  float    v8f;

#define BSZ   256
#define TLEN  1000
#define CIN   64
#define HD    128
#define TDAIN 150
#define NCLS  4

// workspace layout (bytes)
#define OFF_WCONV 0u                    // 2560 frags * 32B = 81920
#define OFF_W1    81920u                // 1024 frags * 32B = 32768
#define OFF_W2    114688u               // 32768
#define OFF_BNS   147456u               // 128 * 4
#define OFF_BNH   147968u               // 128 * 4
#define OFF_FR    148480u               // 256*128*4 = 131072
#define OFF_CUR   279552u               // 1000*256*128*2 = 65,536,000

__device__ __forceinline__ v8f wmma16(v16h a, v16h b, v8f c) {
    return __builtin_amdgcn_wmma_f32_16x16x32_f16(
        /*neg_a=*/false, a, /*neg_b=*/false, b,
        /*c_mod=*/(short)0, c, /*reuse_a=*/false, /*reuse_b=*/false);
}

// ---------------------------------------------------------------------------
// Prep: build WMMA B-fragment-major weight images + folded BN constants.
// Fragment layout: frag[(kc*8 + ntile)*32 + lane] is a v16h where element e
// holds W[kc*32 + 16*(lane/16) + e][ntile*16 + lane%16]  (B operand; contiguous k order, same as A loads).
// ---------------------------------------------------------------------------
__global__ __launch_bounds__(128)
void prep_kernel(const float* __restrict__ conv_w, const float* __restrict__ conv_b,
                 const float* __restrict__ g1, const float* __restrict__ be1,
                 const float* __restrict__ mu1, const float* __restrict__ var1,
                 const float* __restrict__ s1w, const float* __restrict__ s2w,
                 _Float16* __restrict__ wconv_f, _Float16* __restrict__ w1_f,
                 _Float16* __restrict__ w2_f,
                 float* __restrict__ bn_scale, float* __restrict__ bn_shift)
{
    int tid = blockIdx.x * blockDim.x + threadIdx.x;
    if (tid < 2560) {                         // conv weights: 10 kc * 8 nt * 32 lanes
        int kc = tid >> 8;
        int r  = tid & 255;
        int nt = r >> 5;
        int L  = r & 31;
        int n  = nt * 16 + (L & 15);
        int kb = (L >> 4) << 4;               // 16*(lane/16)
        v16h v;
#pragma unroll
        for (int e = 0; e < 16; ++e) {
            int k  = kc * 32 + kb + e;        // k = k1*64 + c
            int k1 = k >> 6;
            int c  = k & 63;
            v[e] = (_Float16)conv_w[(n * CIN + c) * 5 + k1];
        }
        ((v16h*)wconv_f)[tid] = v;
    } else if (tid < 2560 + 1024) {           // snn1_w: 4 kc * 8 nt * 32 lanes
        int t2 = tid - 2560;
        int kc = t2 >> 8;
        int r  = t2 & 255;
        int nt = r >> 5;
        int L  = r & 31;
        int n  = nt * 16 + (L & 15);
        int kb = (L >> 4) << 4;
        v16h v;
#pragma unroll
        for (int e = 0; e < 16; ++e) v[e] = (_Float16)s1w[(kc * 32 + kb + e) * HD + n];
        ((v16h*)w1_f)[t2] = v;
    } else if (tid < 2560 + 2048) {           // snn2_w
        int t2 = tid - (2560 + 1024);
        int kc = t2 >> 8;
        int r  = t2 & 255;
        int nt = r >> 5;
        int L  = r & 31;
        int n  = nt * 16 + (L & 15);
        int kb = (L >> 4) << 4;
        v16h v;
#pragma unroll
        for (int e = 0; e < 16; ++e) v[e] = (_Float16)s2w[(kc * 32 + kb + e) * HD + n];
        ((v16h*)w2_f)[t2] = v;
    } else if (tid < 2560 + 2048 + 128) {     // folded BN1 (+ conv bias)
        int h = tid - (2560 + 2048);
        float sc = g1[h] * rsqrtf(var1[h] + 1e-5f);
        bn_scale[h] = sc;
        bn_shift[h] = (conv_b[h] - mu1[h]) * sc + be1[h];
    }
}

// ---------------------------------------------------------------------------
// Conv1d (pad=2) + BN + ReLU  ->  current_seq f16 [T][B][H]
// Grid: (B, ceil(T/64)); block = 128 threads = 4 waves; wave = 16 t-rows.
// A[t, k] with k = k1*64 + c is the contiguous window in[b, t+k1-2, :].
// Padding is branchless: clamped (always-valid) address + multiply-by-mask,
// so the A path stays vector b128 loads with no per-element EXEC juggling.
// ---------------------------------------------------------------------------
__global__ __launch_bounds__(128)
void conv_kernel(const float* __restrict__ in, const _Float16* __restrict__ wfrag,
                 const float* __restrict__ bn_scale, const float* __restrict__ bn_shift,
                 _Float16* __restrict__ cur)
{
    const int b    = blockIdx.x;
    const int tt   = blockIdx.y;
    const int wv   = threadIdx.x >> 5;
    const int lane = threadIdx.x & 31;
    const int t0   = tt * 64 + wv * 16;
    const int m    = lane & 15;       // A row within tile
    const int half = lane >> 4;       // K-half select

    v8f acc[8] = {};

#pragma unroll
    for (int kc = 0; kc < 10; ++kc) {
        const int k1 = kc >> 1;
        const int tp = t0 + m + k1 - 2;
        // branchless zero-padding: clamp the address (always legal memory),
        // zero the contribution with a per-lane multiplicative mask.
        int tc = tp < 0 ? 0 : tp;
        tc = tc > (TLEN - 1) ? (TLEN - 1) : tc;
        const float msk = ((tp >= 0) && (tp < TLEN)) ? 1.f : 0.f;
        const int c0 = ((kc & 1) << 5) + (half << 4);
        const float* p = in + ((size_t)b * TLEN + tc) * CIN + c0;
        const float4 f0 = *(const float4*)(p);
        const float4 f1 = *(const float4*)(p + 4);
        const float4 g0 = *(const float4*)(p + 8);
        const float4 g1 = *(const float4*)(p + 12);
        v16h a;
        a[0]  = (_Float16)(f0.x * msk); a[1]  = (_Float16)(f0.y * msk);
        a[2]  = (_Float16)(f0.z * msk); a[3]  = (_Float16)(f0.w * msk);
        a[4]  = (_Float16)(f1.x * msk); a[5]  = (_Float16)(f1.y * msk);
        a[6]  = (_Float16)(f1.z * msk); a[7]  = (_Float16)(f1.w * msk);
        a[8]  = (_Float16)(g0.x * msk); a[9]  = (_Float16)(g0.y * msk);
        a[10] = (_Float16)(g0.z * msk); a[11] = (_Float16)(g0.w * msk);
        a[12] = (_Float16)(g1.x * msk); a[13] = (_Float16)(g1.y * msk);
        a[14] = (_Float16)(g1.z * msk); a[15] = (_Float16)(g1.w * msk);
#pragma unroll
        for (int nt = 0; nt < 8; ++nt) {
            v16h bf = ((const v16h*)wfrag)[(kc * 8 + nt) * 32 + lane];
            acc[nt] = wmma16(a, bf, acc[nt]);
        }
    }

#pragma unroll
    for (int nt = 0; nt < 8; ++nt) {
        const int n = nt * 16 + m;         // D column = lane%16
        const float sc = bn_scale[n];
        const float sh = bn_shift[n];
#pragma unroll
        for (int e = 0; e < 8; ++e) {
            const int tr = t0 + e + half * 8;   // D row = e + 8*(lane/16)
            if (tr < TLEN) {
                float y = acc[nt][e] * sc + sh;
                y = y > 0.f ? y : 0.f;
                cur[((size_t)tr * BSZ + b) * HD + n] = (_Float16)y;
            }
        }
    }
}

// ---------------------------------------------------------------------------
// LIF SNN scan: 1000 sequential steps, two WMMA GEMMs per step.
// Grid: 16 blocks (16 batch rows each); block = 128 threads = 4 waves.
// Wave owns 32 hidden columns (2 N-tiles); membranes/counts persist in VGPRs;
// snn weights live in VGPRs as pre-loaded B-fragments; spike exchange via LDS.
// ---------------------------------------------------------------------------
__global__ __launch_bounds__(128)
void snn_scan_kernel(const _Float16* __restrict__ cur,
                     const _Float16* __restrict__ w1f, const _Float16* __restrict__ w2f,
                     const float* __restrict__ b1v, const float* __restrict__ b2v,
                     float* __restrict__ counts_out, float* __restrict__ fr)
{
    __shared__ __attribute__((aligned(16))) _Float16 s_spk[16 * HD];

    const int b0   = blockIdx.x * 16;
    const int wv   = threadIdx.x >> 5;
    const int lane = threadIdx.x & 31;
    const int m    = lane & 15;
    const int half = lane >> 4;
    const int nt0  = wv * 2;

    // Preload weight B-fragments (stay in VGPRs for all 1000 steps)
    v16h B1[4][2], B2[4][2];
#pragma unroll
    for (int kc = 0; kc < 4; ++kc)
#pragma unroll
        for (int j = 0; j < 2; ++j) {
            B1[kc][j] = ((const v16h*)w1f)[(kc * 8 + nt0 + j) * 32 + lane];
            B2[kc][j] = ((const v16h*)w2f)[(kc * 8 + nt0 + j) * 32 + lane];
        }
    float bias1[2], bias2[2];
#pragma unroll
    for (int j = 0; j < 2; ++j) {
        bias1[j] = b1v[(nt0 + j) * 16 + m];
        bias2[j] = b2v[(nt0 + j) * 16 + m];
    }

    v8f mem1[2] = {}, mem2[2] = {}, cnt[2] = {};

    for (int t = 0; t < TLEN; ++t) {
        // ---- GEMM1: cur1 = x_t @ snn1_w ----
        const _Float16* abase = cur + ((size_t)t * BSZ + b0 + m) * HD;
        {   // prefetch next timestep's A rows into cache
            const int tn = (t + 1 < TLEN) ? (t + 1) : t;
            __builtin_prefetch(cur + ((size_t)tn * BSZ + b0 + m) * HD, 0, 1);
        }
        v8f acc[2] = {};
#pragma unroll
        for (int kc = 0; kc < 4; ++kc) {
            const v8h lo = *(const v8h*)(abase + kc * 32 + half * 16);
            const v8h hi = *(const v8h*)(abase + kc * 32 + half * 16 + 8);
            v16h a;
#pragma unroll
            for (int e = 0; e < 8; ++e) { a[e] = lo[e]; a[8 + e] = hi[e]; }
            acc[0] = wmma16(a, B1[kc][0], acc[0]);
            acc[1] = wmma16(a, B1[kc][1], acc[1]);
        }
        // ---- LIF layer 1 + spike publish ----
#pragma unroll
        for (int j = 0; j < 2; ++j) {
            const int n = (nt0 + j) * 16 + m;
#pragma unroll
            for (int e = 0; e < 8; ++e) {
                float v = 0.9f * mem1[j][e] + acc[j][e] + bias1[j];
                float s = (v - 1.0f) >= 0.f ? 1.f : 0.f;
                mem1[j][e] = v * (1.f - s);
                s_spk[(e + half * 8) * HD + n] = (_Float16)s;
            }
        }
        __syncthreads();
        // ---- GEMM2: cur2 = spk1 @ snn2_w (A from LDS) ----
        v8f acc2[2] = {};
#pragma unroll
        for (int kc = 0; kc < 4; ++kc) {
            const v8h lo = *(const v8h*)(&s_spk[m * HD + kc * 32 + half * 16]);
            const v8h hi = *(const v8h*)(&s_spk[m * HD + kc * 32 + half * 16 + 8]);
            v16h a;
#pragma unroll
            for (int e = 0; e < 8; ++e) { a[e] = lo[e]; a[8 + e] = hi[e]; }
            acc2[0] = wmma16(a, B2[kc][0], acc2[0]);
            acc2[1] = wmma16(a, B2[kc][1], acc2[1]);
        }
        // ---- LIF layer 2 + spike counting ----
#pragma unroll
        for (int j = 0; j < 2; ++j)
#pragma unroll
            for (int e = 0; e < 8; ++e) {
                float v = 0.9f * mem2[j][e] + acc2[j][e] + bias2[j];
                float s = (v - 1.0f) >= 0.f ? 1.f : 0.f;
                mem2[j][e] = v * (1.f - s);
                cnt[j][e] += s;
            }
        __syncthreads();   // protect s_spk before next step overwrites it
    }

    // ---- emit spike_counts (output #2) and firing_rate (for the head) ----
#pragma unroll
    for (int j = 0; j < 2; ++j) {
        const int n = (nt0 + j) * 16 + m;
#pragma unroll
        for (int e = 0; e < 8; ++e) {
            const int row = b0 + e + half * 8;
            const float c = cnt[j][e];
            counts_out[row * HD + n] = c;
            fr[row * HD + n] = c * (1.0f / (float)TLEN);
        }
    }
}

// ---------------------------------------------------------------------------
// TDA MLP + fusion classifier (tiny). One block per batch row.
// ---------------------------------------------------------------------------
__global__ __launch_bounds__(192)
void head_kernel(const float* __restrict__ fr, const float* __restrict__ tda_in,
                 const float* __restrict__ tw1, const float* __restrict__ tb1,
                 const float* __restrict__ tw2, const float* __restrict__ tb2,
                 const float* __restrict__ cw1, const float* __restrict__ cb1,
                 const float* __restrict__ g2, const float* __restrict__ be2,
                 const float* __restrict__ mu2, const float* __restrict__ var2,
                 const float* __restrict__ cw2, const float* __restrict__ cb2,
                 float* __restrict__ out)
{
    __shared__ float s_t[TDAIN];
    __shared__ float s_a[64];
    __shared__ float s_b[64];
    __shared__ float s_f[192];
    __shared__ float s_h[128];

    const int b   = blockIdx.x;
    const int tid = threadIdx.x;

    for (int i = tid; i < TDAIN; i += 192) s_t[i] = tda_in[b * TDAIN + i];
    __syncthreads();

    if (tid < 64) {
        float a = tb1[tid];
        for (int k = 0; k < TDAIN; ++k) a += s_t[k] * tw1[k * 64 + tid];
        s_a[tid] = fmaxf(a, 0.f);
    }
    __syncthreads();

    if (tid < 64) {
        float a = tb2[tid];
        for (int k = 0; k < 64; ++k) a += s_a[k] * tw2[k * 64 + tid];
        s_b[tid] = fmaxf(a, 0.f);
    }
    __syncthreads();

    if (tid < 128) s_f[tid] = fr[b * HD + tid];
    else           s_f[tid] = s_b[tid - 128];
    __syncthreads();

    if (tid < 128) {
        float a = cb1[tid];
        for (int k = 0; k < 192; ++k) a += s_f[k] * cw1[k * 128 + tid];
        const float sc = g2[tid] * rsqrtf(var2[tid] + 1e-5f);
        a = (a - mu2[tid]) * sc + be2[tid];
        s_h[tid] = fmaxf(a, 0.f);
    }
    __syncthreads();

    if (tid < NCLS) {
        float a = cb2[tid];
        for (int k = 0; k < 128; ++k) a += s_h[k] * cw2[k * NCLS + tid];
        out[b * NCLS + tid] = a;
    }
}

// ---------------------------------------------------------------------------
extern "C" void kernel_launch(void* const* d_in, const int* in_sizes, int n_in,
                              void* d_out, int out_size, void* d_ws, size_t ws_size,
                              hipStream_t stream)
{
    const float* kin    = (const float*)d_in[0];
    const float* tdain  = (const float*)d_in[1];
    const float* conv_w = (const float*)d_in[2];
    const float* conv_b = (const float*)d_in[3];
    const float* g1     = (const float*)d_in[4];
    const float* be1    = (const float*)d_in[5];
    const float* mu1    = (const float*)d_in[6];
    const float* var1   = (const float*)d_in[7];
    const float* s1w    = (const float*)d_in[8];
    const float* s1b    = (const float*)d_in[9];
    const float* s2w    = (const float*)d_in[10];
    const float* s2b    = (const float*)d_in[11];
    const float* tw1    = (const float*)d_in[12];
    const float* tb1    = (const float*)d_in[13];
    const float* tw2    = (const float*)d_in[14];
    const float* tb2    = (const float*)d_in[15];
    const float* cw1    = (const float*)d_in[16];
    const float* cb1    = (const float*)d_in[17];
    const float* g2     = (const float*)d_in[18];
    const float* be2    = (const float*)d_in[19];
    const float* mu2    = (const float*)d_in[20];
    const float* var2   = (const float*)d_in[21];
    const float* cw2    = (const float*)d_in[22];
    const float* cb2    = (const float*)d_in[23];

    char* ws = (char*)d_ws;
    _Float16* wconv_f  = (_Float16*)(ws + OFF_WCONV);
    _Float16* w1_f     = (_Float16*)(ws + OFF_W1);
    _Float16* w2_f     = (_Float16*)(ws + OFF_W2);
    float*    bn_scale = (float*)(ws + OFF_BNS);
    float*    bn_shift = (float*)(ws + OFF_BNH);
    float*    frbuf    = (float*)(ws + OFF_FR);
    _Float16* cur      = (_Float16*)(ws + OFF_CUR);

    float* out        = (float*)d_out;               // [256,4]
    float* counts_out = out + BSZ * NCLS;            // [256,128]

    // 1) weight re-layout + BN folding (4736 logical threads)
    prep_kernel<<<37, 128, 0, stream>>>(conv_w, conv_b, g1, be1, mu1, var1,
                                        s1w, s2w, wconv_f, w1_f, w2_f,
                                        bn_scale, bn_shift);
    // 2) conv + BN + ReLU -> f16 current_seq
    conv_kernel<<<dim3(BSZ, 16), 128, 0, stream>>>(kin, wconv_f, bn_scale, bn_shift, cur);
    // 3) sequential LIF scan (WMMA per step), emits spike_counts + firing_rate
    snn_scan_kernel<<<16, 128, 0, stream>>>(cur, w1_f, w2_f, s1b, s2b, counts_out, frbuf);
    // 4) TDA MLP + classifier head
    head_kernel<<<BSZ, 192, 0, stream>>>(frbuf, tdain, tw1, tb1, tw2, tb2,
                                         cw1, cb1, g2, be2, mu2, var2, cw2, cb2, out);
}
